// MultiHeadSelfAttention_5111011082273
// MI455X (gfx1250) — matrix operand order repaired and checked by reading, not compiled, not run on hardware
//
#include <hip/hip_runtime.h>
#include <hip/hip_bf16.h>

// ---------------------------------------------------------------------------
// MHSA for B=4, S=2048, D=768, H=12, Dh=64 on gfx1250 (CDNA5, wave32, WMMA)
// ---------------------------------------------------------------------------

typedef __attribute__((ext_vector_type(16))) _Float16 v16h;
typedef __attribute__((ext_vector_type(8)))  _Float16 v8h;
typedef __attribute__((ext_vector_type(4)))  _Float16 v4h;
typedef __attribute__((ext_vector_type(8)))  float    v8f;

#define WMMA_F16(a, b, c) \
  __builtin_amdgcn_wmma_f32_16x16x32_f16(false, (a), false, (b), (short)0, (c), false, false)

constexpr int Bn  = 4;
constexpr int S   = 2048;
constexpr int D   = 768;
constexpr int H   = 12;
constexpr int Dh  = 64;
constexpr int BS  = Bn * S;          // 8192 rows
constexpr int HS_ELEMS = BS * D;     // 6291456 (= B*H*S*Dh too)
constexpr int W_ELEMS  = D * D;      // 589824

// ---------------------------------------------------------------------------
// Fragment loaders in contiguous k order (not the ISA table order; A and B use the same).
// A (16x32 f16): lane m   -> K = k0+0..15
//                lane m+16-> K = k0+16..31                   (row M = lane&15)
// B (32x16 f16): lanes 0-15  hold K = k0+0..15  (col N = lane&15)
//                lanes 16-31 hold K = k0+16..31
// ---------------------------------------------------------------------------
__device__ inline v16h load_a_frag(const _Float16* __restrict__ base, int ldm,
                                   int k0, int lane) {
  const int m  = lane & 15;
  const int hi = (lane >> 4) & 1;
  const _Float16* p = base + (size_t)m * ldm + k0 + hi * 16;
  v8h lo = *(const v8h*)(p);        // K = k0 + hi*16 + [0,8)
  v8h up = *(const v8h*)(p + 8);    // K = k0 + hi*16 + 8 + [0,8)
  v16h r;
#pragma unroll
  for (int i = 0; i < 8; ++i) { r[i] = lo[i]; r[i + 8] = up[i]; }
  return r;
}

__device__ inline v16h load_b_frag(const _Float16* __restrict__ base, int ldn,
                                   int k0, int lane) {
  const int n  = lane & 15;
  const int hi = (lane >> 4) & 1;
  return *(const v16h*)(base + (size_t)n * ldn + k0 + hi * 16);
}

// ---------------------------------------------------------------------------
// Kernel 0: f32 -> f16 conversion (hidden states + weights), vectorized x4
// ---------------------------------------------------------------------------
__global__ __launch_bounds__(256) void cvt_f32_f16(const float* __restrict__ src,
                                                   _Float16* __restrict__ dst,
                                                   int n4) {
  int i = blockIdx.x * blockDim.x + threadIdx.x;
  if (i < n4) {
    float4 v = ((const float4*)src)[i];
    v4h o;
    o[0] = (_Float16)v.x; o[1] = (_Float16)v.y;
    o[2] = (_Float16)v.z; o[3] = (_Float16)v.w;
    ((v4h*)dst)[i] = o;
  }
}

// ---------------------------------------------------------------------------
// Kernel 1: fused QKV projection + RoPE.
//   grid = (BS/64, H, 3{q,k,v}), block = 128 (4 waves).
//   Each block computes a 64-row x 64-col (one head) tile of X @ W^T.
//   Q gets RoPE and the 1/sqrt(Dh) scale; K gets RoPE; V is stored
//   TRANSPOSED as vT[b,h,dh,s] so attention's P@V B-frags are contiguous.
// ---------------------------------------------------------------------------
__global__ __launch_bounds__(128) void qkv_rope_kernel(
    const _Float16* __restrict__ h16,
    const _Float16* __restrict__ wq, const _Float16* __restrict__ wk,
    const _Float16* __restrict__ wv,
    _Float16* __restrict__ q16, _Float16* __restrict__ k16,
    _Float16* __restrict__ vT16) {
  const int which = blockIdx.z;
  const int h     = blockIdx.y;
  const int Mbase = blockIdx.x * 64;
  const int lane  = threadIdx.x & 31;
  const int wave  = threadIdx.x >> 5;
  const int rowbase = Mbase + wave * 16;

  const _Float16* W = (which == 0) ? wq : (which == 1) ? wk : wv;
  const _Float16* Arow = h16 + (size_t)rowbase * D;
  const _Float16* Brow = W + (size_t)(h * Dh) * D;

  v8f acc[4];
#pragma unroll
  for (int j = 0; j < 4; ++j)
#pragma unroll
    for (int i = 0; i < 8; ++i) acc[j][i] = 0.0f;

  for (int k0 = 0; k0 < D; k0 += 32) {
    v16h a = load_a_frag(Arow, D, k0, lane);
#pragma unroll
    for (int j = 0; j < 4; ++j) {
      v16h b = load_b_frag(Brow + (size_t)(j * 16) * D, D, k0, lane);
      acc[j] = WMMA_F16(a, b, acc[j]);
    }
  }

  const int n   = lane & 15;
  const int rlo = (lane < 16) ? 0 : 8;  // C layout: lanes 16-31 hold rows M=8..15

  if (which < 2) {
    // RoPE: out[dh]    = -y[dh+32]*cos + y[dh]*sin
    //       out[dh+32] =  y[dh]   *cos + y[dh+32]*sin,  angle = s * 10000^(-dh/32)
    const float qscale = (which == 0) ? 0.125f : 1.0f;  // 1/sqrt(64) folded into Q
    _Float16* dst = (which == 0) ? q16 : k16;
#pragma unroll
    for (int j = 0; j < 2; ++j) {
      const int dh = j * 16 + n;  // [0,32)
      const float invfreq = __powf(10000.0f, -(float)dh * (1.0f / 32.0f));
#pragma unroll
      for (int r = 0; r < 8; ++r) {
        const int row = rowbase + rlo + r;
        const int b   = row >> 11;       // /2048
        const int s   = row & 2047;
        float sn, cs;
        __sincosf((float)s * invfreq, &sn, &cs);
        const float ya = acc[j][r], yb = acc[j + 2][r];
        const float oa = (-yb * cs + ya * sn) * qscale;
        const float ob = ( ya * cs + yb * sn) * qscale;
        const size_t base = (((size_t)(b * H + h) * S) + s) * Dh;
        dst[base + dh]      = (_Float16)oa;
        dst[base + dh + 32] = (_Float16)ob;
      }
    }
  } else {
    // V: store transposed vT[b,h,dh,s]; 8 consecutive rows pack to one 16B store
#pragma unroll
    for (int j = 0; j < 4; ++j) {
      const int dh   = j * 16 + n;
      const int row0 = rowbase + rlo;
      const int b    = row0 >> 11;
      const int s0   = row0 & 2047;
      v8h pack;
#pragma unroll
      for (int r = 0; r < 8; ++r) pack[r] = (_Float16)acc[j][r];
      *(v8h*)(vT16 + (((size_t)(b * H + h) * Dh + dh) * S) + s0) = pack;
    }
  }
}

// ---------------------------------------------------------------------------
// Kernel 2: flash attention per (b,h,q-tile of 64). block = 128 (4 waves,
// each wave owns 16 query rows; no cross-wave sharing -> no barriers).
// ---------------------------------------------------------------------------
__global__ __launch_bounds__(128) void attn_kernel(
    const _Float16* __restrict__ q16, const _Float16* __restrict__ k16,
    const _Float16* __restrict__ vT16, const float* __restrict__ mask,
    _Float16* __restrict__ ctx16) {
  const int bh   = blockIdx.y;
  const int b    = bh / H;
  const int h    = bh % H;
  const int lane = threadIdx.x & 31;
  const int wave = threadIdx.x >> 5;
  const int qrow = blockIdx.x * 64 + wave * 16;
  const int n    = lane & 15;
  const int rlo  = (lane < 16) ? 0 : 8;

  const _Float16* Q  = q16  + (size_t)bh * S * Dh;
  const _Float16* K  = k16  + (size_t)bh * S * Dh;
  const _Float16* V  = vT16 + (size_t)bh * Dh * S;  // [dh][s]
  const float* maskb = mask + (size_t)b * S;

  // Q A-fragments for the two 32-wide K-dim steps (Dh = 64)
  const v16h aq0 = load_a_frag(Q + (size_t)qrow * Dh, Dh, 0,  lane);
  const v16h aq1 = load_a_frag(Q + (size_t)qrow * Dh, Dh, 32, lane);

  __shared__ _Float16 Ps[4][16][64];  // per-wave P staging (transpose via LDS)
  _Float16 (*myPs)[64] = Ps[wave];

  float mrow[8], lrow[8];
  v8f ctx[4];
#pragma unroll
  for (int r = 0; r < 8; ++r) { mrow[r] = -1e30f; lrow[r] = 0.0f; }
#pragma unroll
  for (int j = 0; j < 4; ++j)
#pragma unroll
    for (int i = 0; i < 8; ++i) ctx[j][i] = 0.0f;

  for (int kt = 0; kt < S / 64; ++kt) {
    const int kbase = kt * 64;
    if (kt + 1 < S / 64) {  // hint next tiles toward the caches
      __builtin_prefetch((const void*)(K + (size_t)(kbase + 64) * Dh), 0, 3);
      __builtin_prefetch((const void*)(V + kbase + 64), 0, 3);
    }

    // ---- scores S = Q @ K^T  (16 x 64) ----
    v8f sc[4];
#pragma unroll
    for (int j = 0; j < 4; ++j) {
#pragma unroll
      for (int i = 0; i < 8; ++i) sc[j][i] = 0.0f;
      const _Float16* Krow = K + (size_t)(kbase + j * 16) * Dh;
      sc[j] = WMMA_F16(aq0, load_b_frag(Krow, Dh, 0,  lane), sc[j]);
      sc[j] = WMMA_F16(aq1, load_b_frag(Krow, Dh, 32, lane), sc[j]);
    }

    // ---- additive mask ----
#pragma unroll
    for (int j = 0; j < 4; ++j) {
      const float em = (1.0f - maskb[kbase + j * 16 + n]) * -10000.0f;
#pragma unroll
      for (int r = 0; r < 8; ++r) sc[j][r] += em;
    }

    // ---- online softmax (row stats live in 16-lane groups) ----
    float mx[8];
#pragma unroll
    for (int r = 0; r < 8; ++r) {
      float m = sc[0][r];
      m = fmaxf(m, sc[1][r]); m = fmaxf(m, sc[2][r]); m = fmaxf(m, sc[3][r]);
#pragma unroll
      for (int off = 1; off < 16; off <<= 1)
        m = fmaxf(m, __shfl_xor(m, off, 32));
      mx[r] = m;
    }
    float scale[8];
#pragma unroll
    for (int r = 0; r < 8; ++r) {
      const float mn = fmaxf(mrow[r], mx[r]);
      scale[r] = __expf(mrow[r] - mn);
      mrow[r]  = mn;
    }
    float rs[8];
#pragma unroll
    for (int r = 0; r < 8; ++r) rs[r] = 0.0f;
#pragma unroll
    for (int j = 0; j < 4; ++j)
#pragma unroll
      for (int r = 0; r < 8; ++r) {
        const float p = __expf(sc[j][r] - mrow[r]);
        sc[j][r] = p;
        rs[r] += p;
      }
#pragma unroll
    for (int r = 0; r < 8; ++r) {
#pragma unroll
      for (int off = 1; off < 16; off <<= 1)
        rs[r] += __shfl_xor(rs[r], off, 32);
      lrow[r] = lrow[r] * scale[r] + rs[r];
    }
#pragma unroll
    for (int j = 0; j < 4; ++j)
#pragma unroll
      for (int r = 0; r < 8; ++r) ctx[j][r] *= scale[r];

    // ---- P -> LDS (f16) to re-enter WMMA as A-fragment ----
#pragma unroll
    for (int j = 0; j < 4; ++j)
#pragma unroll
      for (int r = 0; r < 8; ++r)
        myPs[rlo + r][j * 16 + n] = (_Float16)sc[j][r];

    // ---- ctx += P @ V  (V pre-transposed: row dh, contiguous keys) ----
#pragma unroll
    for (int kk = 0; kk < 2; ++kk) {
      v16h ap = load_a_frag(&myPs[0][0], 64, kk * 32, lane);
#pragma unroll
      for (int j = 0; j < 4; ++j) {
        v16h bv = load_b_frag(V + (size_t)(j * 16) * S, S, kbase + kk * 32, lane);
        ctx[j] = WMMA_F16(ap, bv, ctx[j]);
      }
    }
  }

  // ---- normalize and store ctx16[b, s, h*64+dh] ----
#pragma unroll
  for (int j = 0; j < 4; ++j)
#pragma unroll
    for (int r = 0; r < 8; ++r) {
      const int srow = qrow + rlo + r;
      const float o  = ctx[j][r] / lrow[r];
      ctx16[((size_t)b * S + srow) * D + h * Dh + j * 16 + n] = (_Float16)o;
    }
}

// ---------------------------------------------------------------------------
// Kernel 3: output projection out = ctx @ Wo^T + bo  (f32 output)
//   grid = (BS/64, D/64), block = 128 (4 waves).
// ---------------------------------------------------------------------------
__global__ __launch_bounds__(128) void outproj_kernel(
    const _Float16* __restrict__ ctx16, const _Float16* __restrict__ wo,
    const float* __restrict__ bo, float* __restrict__ out) {
  const int Mbase = blockIdx.x * 64;
  const int Nbase = blockIdx.y * 64;
  const int lane  = threadIdx.x & 31;
  const int wave  = threadIdx.x >> 5;
  const int rowbase = Mbase + wave * 16;
  const int n   = lane & 15;
  const int rlo = (lane < 16) ? 0 : 8;

  const _Float16* Arow = ctx16 + (size_t)rowbase * D;
  const _Float16* Brow = wo + (size_t)Nbase * D;

  v8f acc[4];
#pragma unroll
  for (int j = 0; j < 4; ++j)
#pragma unroll
    for (int i = 0; i < 8; ++i) acc[j][i] = 0.0f;

  for (int k0 = 0; k0 < D; k0 += 32) {
    v16h a = load_a_frag(Arow, D, k0, lane);
#pragma unroll
    for (int j = 0; j < 4; ++j) {
      v16h b = load_b_frag(Brow + (size_t)(j * 16) * D, D, k0, lane);
      acc[j] = WMMA_F16(a, b, acc[j]);
    }
  }

#pragma unroll
  for (int j = 0; j < 4; ++j) {
    const int col = Nbase + j * 16 + n;
    const float bias = bo[col];
#pragma unroll
    for (int r = 0; r < 8; ++r) {
      const int row = rowbase + rlo + r;
      out[(size_t)row * D + col] = acc[j][r] + bias;
    }
  }
}

// ---------------------------------------------------------------------------
// Host launcher
// ---------------------------------------------------------------------------
extern "C" void kernel_launch(void* const* d_in, const int* in_sizes, int n_in,
                              void* d_out, int out_size, void* d_ws, size_t ws_size,
                              hipStream_t stream) {
  const float* hidden = (const float*)d_in[0];
  const float* amask  = (const float*)d_in[1];
  const float* Wq     = (const float*)d_in[2];
  const float* Wk     = (const float*)d_in[3];
  const float* Wv     = (const float*)d_in[4];
  const float* Wo     = (const float*)d_in[5];
  const float* bo     = (const float*)d_in[6];
  float* out          = (float*)d_out;

  _Float16* ws   = (_Float16*)d_ws;
  _Float16* h16  = ws;
  _Float16* wq16 = h16  + (size_t)HS_ELEMS;
  _Float16* wk16 = wq16 + (size_t)W_ELEMS;
  _Float16* wv16 = wk16 + (size_t)W_ELEMS;
  _Float16* wo16 = wv16 + (size_t)W_ELEMS;
  _Float16* q16  = wo16 + (size_t)W_ELEMS;
  _Float16* k16  = q16  + (size_t)HS_ELEMS;
  _Float16* vT16 = k16  + (size_t)HS_ELEMS;
  _Float16* c16  = vT16 + (size_t)HS_ELEMS;

  // 0) convert inputs to f16
  {
    int n4 = HS_ELEMS / 4;
    cvt_f32_f16<<<(n4 + 255) / 256, 256, 0, stream>>>(hidden, h16, n4);
    n4 = W_ELEMS / 4;
    cvt_f32_f16<<<(n4 + 255) / 256, 256, 0, stream>>>(Wq, wq16, n4);
    cvt_f32_f16<<<(n4 + 255) / 256, 256, 0, stream>>>(Wk, wk16, n4);
    cvt_f32_f16<<<(n4 + 255) / 256, 256, 0, stream>>>(Wv, wv16, n4);
    cvt_f32_f16<<<(n4 + 255) / 256, 256, 0, stream>>>(Wo, wo16, n4);
  }

  // 1) fused QKV + RoPE (Q pre-scaled by 1/sqrt(Dh); V stored transposed)
  qkv_rope_kernel<<<dim3(BS / 64, H, 3), 128, 0, stream>>>(
      h16, wq16, wk16, wv16, q16, k16, vT16);

  // 2) flash attention
  attn_kernel<<<dim3(S / 64, Bn * H), 128, 0, stream>>>(
      q16, k16, vT16, amask, c16);

  // 3) output projection + bias
  outproj_kernel<<<dim3(BS / 64, D / 64), 128, 0, stream>>>(
      c16, wo16, bo, out);
}
